// BatchHardContrastiveLoss_56478819942996
// MI455X (gfx1250) — matrix operand order repaired and checked by reading, not compiled, not run on hardware
//
#include <hip/hip_runtime.h>
#include <hip/hip_bf16.h>

// BatchHardContrastiveLoss, N=8192, D=256.
// init -> f32->f16 + norms -> WMMA Gram tiles with fused squared-distance
// mining (async global->LDS B staging, mask bytes software-pipelined one
// tile ahead, sqrt deferred to per-row pass) -> row losses -> scalar.

#define NN 8192
#define DD 256
#define NTILES (NN / 16)       // 512 column tiles of 16
#define CTILES_PER_BLK 64      // column tiles per block (grid.y = 8)

#define POS_SENT -3.0e38f
#define NEG_SENT  3.0e38f

typedef _Float16 v8h  __attribute__((ext_vector_type(8)));
typedef _Float16 v16h __attribute__((ext_vector_type(16)));
typedef float    v8f  __attribute__((ext_vector_type(8)));

// ---------------------------------------------------------------- init ----
__global__ __launch_bounds__(256) void bhc_init(int* __restrict__ gpos,
                                                int* __restrict__ gneg,
                                                float* __restrict__ acc) {
  const int i = blockIdx.x * 256 + threadIdx.x;
  if (i < NN) {
    gpos[i] = 0;                            // bits of 0.0f; any d2 >= 1e-12
    gneg[i] = __float_as_int(NEG_SENT);     // int order == float order (>=0)
  }
  if (i < 4) acc[i] = 0.0f;                 // [sumP, cntP, sumN, cntN]
}

// ------------------------------------------------- f32 -> f16 + norms ----
__global__ __launch_bounds__(256) void bhc_prep(const float* __restrict__ x,
                                                _Float16* __restrict__ xh,
                                                float* __restrict__ sq) {
  const int row = blockIdx.x;     // 8192 blocks
  const int t   = threadIdx.x;    // 256 threads = D
  const float v = x[(size_t)row * DD + t];
  xh[(size_t)row * DD + t] = (_Float16)v;
  __shared__ float red[256];
  red[t] = v * v;
  __syncthreads();
  for (int s = 128; s > 0; s >>= 1) {
    if (t < s) red[t] += red[t + s];
    __syncthreads();
  }
  if (t == 0) sq[row] = red[0];
}

// ------------------------------------------------------------- main ------
// grid = (64, 8): blockIdx.x selects 128 rows (16 per wave, 8 waves),
// blockIdx.y selects a chunk of 64 column tiles. Per tile: async B copy for
// it+1, mask/sq loads for it+1 (consumed next iteration), 16 ds_load_b128,
// sched_barrier, 8 chained WMMAs, compare/select mining on preloaded masks.
__global__ __launch_bounds__(256) void bhc_main(
    const _Float16* __restrict__ xh, const float* __restrict__ sq,
    const unsigned char* __restrict__ pmask,
    const unsigned char* __restrict__ nmask,
    int* __restrict__ gpos, int* __restrict__ gneg) {
  __shared__ __align__(16) _Float16 Bt[2][16 * DD];   // 2 x 8KB

  const int t    = threadIdx.x;
  const int wave = t >> 5;
  const int lane = t & 31;
  const int l15  = lane & 15;
  const int h    = lane >> 4;                 // lane half (0/1)

  const int i0  = blockIdx.x * 128 + wave * 16;   // rows owned by this wave
  const int jt0 = blockIdx.y * CTILES_PER_BLK;

  // ---- A tile 16x256 in WMMA A layout (ISA 7.12.2, 16-bit 16x32) ----
  v16h a[8];
  {
    const _Float16* arow = xh + (size_t)(i0 + l15) * DD;
#pragma unroll
    for (int ks = 0; ks < 8; ++ks) {
      v8h lo = *(const v8h*)(arow + ks * 32 + h * 8);
      v8h hi = *(const v8h*)(arow + ks * 32 + 16 + h * 8);
      a[ks] = __builtin_shufflevector(lo, hi, 0, 1, 2, 3, 4, 5, 6, 7,
                                      8, 9, 10, 11, 12, 13, 14, 15);
    }
  }

  float pos_run[8], neg_run[8];   // running max/min of (sq_j - 2*dot)
#pragma unroll
  for (int r = 0; r < 8; ++r) { pos_run[r] = POS_SENT; neg_run[r] = NEG_SENT; }

  // async cooperative copy of one B tile (16 rows x 256 f16 = 8KB) -> LDS.
  auto loadB_async = [&](int jt, int buf) {
    const int j0 = jt * 16;
#pragma unroll
    for (int c = t; c < 512; c += 256) {
      const int row = c >> 5;
      const int col = (c & 31) * 8;
      const _Float16* gsrc = xh + (size_t)(j0 + row) * DD + col;
      unsigned ldsoff = (unsigned)(size_t)&Bt[buf][row * DD + col];
      asm volatile("global_load_async_to_lds_b128 %0, %1, off"
                   :: "v"(ldsoff), "v"(gsrc)
                   : "memory");
    }
  };

  // mask/sq pipeline: running per-lane pointers, advanced +16 cols per tile
  const unsigned char* pb  = pmask + (size_t)(i0 + 8 * h) * NN + jt0 * 16 + l15;
  const unsigned char* nb  = nmask + (size_t)(i0 + 8 * h) * NN + jt0 * 16 + l15;
  const float*         sqp = sq + jt0 * 16 + l15;

  unsigned pcur[8], ncur[8], pnxt[8], nnxt[8];
  float sqc, sqn;
  auto loadMasks = [&](unsigned (&p)[8], unsigned (&n)[8], float& s) {
#pragma unroll
    for (int r = 0; r < 8; ++r) {
      p[r] = pb[(size_t)r * NN];
      n[r] = nb[(size_t)r * NN];
    }
    s = *sqp;
    pb += 16; nb += 16; sqp += 16;
  };

  loadB_async(jt0, 0);
  loadMasks(pcur, ncur, sqc);
  asm volatile("s_wait_asynccnt 0" ::: "memory");
  __syncthreads();

  for (int it = 0; it < CTILES_PER_BLK; ++it) {
    const int buf = it & 1;
    if (it + 1 < CTILES_PER_BLK) {
      loadB_async(jt0 + it + 1, buf ^ 1);     // LDS copy for next tile
      loadMasks(pnxt, nnxt, sqn);             // HBM mask bytes for next tile
    }

    // all 16 ds_load_b128 first ...
    v16h b[8];
    {
      const _Float16* brow = &Bt[buf][l15 * DD];
#pragma unroll
      for (int ks = 0; ks < 8; ++ks) {
        v8h lo = *(const v8h*)(brow + ks * 32 + h * 8);
        v8h hi = *(const v8h*)(brow + ks * 32 + 16 + h * 8);
        b[ks] = __builtin_shufflevector(lo, hi, 0, 1, 2, 3, 4, 5, 6, 7,
                                        8, 9, 10, 11, 12, 13, 14, 15);
      }
    }
    __builtin_amdgcn_sched_barrier(0);
    // ... then the WMMA chain with graduated DS waits
    v8f cacc = {};
#pragma unroll
    for (int ks = 0; ks < 8; ++ks)
      cacc = __builtin_amdgcn_wmma_f32_16x16x32_f16(
          false, a[ks], false, b[ks], (short)0, cacc, false, false);
    __builtin_amdgcn_sched_barrier(0);

    // mining on val = sq_j - 2*dot (monotone in d^2 per row); masks were
    // preloaded last iteration -> no memory latency on this path.
#pragma unroll
    for (int r = 0; r < 8; ++r) {
      const float val = fmaf(-2.0f, cacc[r], sqc);
      if (pcur[r] && val > pos_run[r]) pos_run[r] = val;
      if (ncur[r] && val < neg_run[r]) neg_run[r] = val;
    }

    if (it + 1 < CTILES_PER_BLK) {
#pragma unroll
      for (int r = 0; r < 8; ++r) { pcur[r] = pnxt[r]; ncur[r] = nnxt[r]; }
      sqc = sqn;
    }

    asm volatile("s_wait_asynccnt 0" ::: "memory");  // our copies landed
    __syncthreads();                                 // everyone's copies done
  }

  // reduce across the 16 lanes sharing each row (groups of 16 within wave32)
#pragma unroll
  for (int r = 0; r < 8; ++r) {
#pragma unroll
    for (int s = 1; s < 16; s <<= 1) {
      pos_run[r] = fmaxf(pos_run[r], __shfl_xor(pos_run[r], s, 16));
      neg_run[r] = fminf(neg_run[r], __shfl_xor(neg_run[r], s, 16));
    }
  }
  if (l15 == 0) {
#pragma unroll
    for (int r = 0; r < 8; ++r) {
      const int   i   = i0 + 8 * h + r;
      const float sqi = sq[i];
      if (pos_run[r] > -1.0e38f) {            // saw >=1 positive in chunk
        const float p2 = fmaxf(sqi + pos_run[r], 1e-12f);   // clamped d^2
        atomicMax(&gpos[i], __float_as_int(p2));            // >=0: int order ok
      }
      if (neg_run[r] < 1.0e38f) {             // saw >=1 negative in chunk
        const float n2 = fmaxf(sqi + neg_run[r], 1e-12f);
        atomicMin(&gneg[i], __float_as_int(n2));
      }
    }
  }
}

// --------------------------------------------------------- row losses ----
__global__ __launch_bounds__(256) void bhc_rowloss(const int* __restrict__ gpos,
                                                   const int* __restrict__ gneg,
                                                   float* __restrict__ acc) {
  const int i = blockIdx.x * 256 + threadIdx.x;
  if (i >= NN) return;
  const int   pb2 = gpos[i];
  const float n2  = __int_as_float(gneg[i]);
  const bool has_pos = pb2 != 0;       // untouched init means no positive
  const bool has_neg = n2 < 1.0e37f;   // untouched sentinel means no negative
  const bool valid   = has_pos && has_neg;
  const float hp = __builtin_sqrtf(__int_as_float(pb2));  // d^2 -> distance
  const float hn = __builtin_sqrtf(n2);
  const float pl = valid ? fmaxf(hp - 0.2f, 0.0f) : 0.0f;
  const float nl = valid ? fmaxf(0.2f - hn, 0.0f) : 0.0f;
  if (pl > 0.0f) { atomicAdd(&acc[0], pl); atomicAdd(&acc[1], 1.0f); }
  if (nl > 0.0f) { atomicAdd(&acc[2], nl); atomicAdd(&acc[3], 1.0f); }
}

// ------------------------------------------------------------ finalize ---
__global__ void bhc_finalize(const float* __restrict__ acc,
                             float* __restrict__ out) {
  if (threadIdx.x == 0 && blockIdx.x == 0) {
    const float p = acc[1] > 0.0f ? acc[0] / acc[1] : 0.0f;
    const float n = acc[3] > 0.0f ? acc[2] / acc[3] : 0.0f;
    out[0] = p + n;
  }
}

// ------------------------------------------------------------- launch ----
extern "C" void kernel_launch(void* const* d_in, const int* in_sizes, int n_in,
                              void* d_out, int out_size, void* d_ws,
                              size_t ws_size, hipStream_t stream) {
  const float*         x  = (const float*)d_in[0];
  const unsigned char* pm = (const unsigned char*)d_in[1];  // bool = 1 byte
  const unsigned char* nm = (const unsigned char*)d_in[2];
  float* out = (float*)d_out;

  // workspace layout (~4.3 MB)
  char*     ws   = (char*)d_ws;
  _Float16* xh   = (_Float16*)ws;                                   // 4 MB
  float*    sq   = (float*)(ws + (size_t)NN * DD * 2);              // 32 KB
  int*      gpos = (int*)(ws + (size_t)NN * DD * 2 + (size_t)NN * 4);
  int*      gneg = gpos + NN;
  float*    acc  = (float*)(gneg + NN);

  bhc_init<<<NN / 256, 256, 0, stream>>>(gpos, gneg, acc);
  bhc_prep<<<NN, 256, 0, stream>>>(x, xh, sq);
  dim3 grid(NN / 128, NTILES / CTILES_PER_BLK);   // (64, 8)
  bhc_main<<<grid, 256, 0, stream>>>(xh, sq, pm, nm, gpos, gneg);
  bhc_rowloss<<<NN / 256, 256, 0, stream>>>(gpos, gneg, acc);
  bhc_finalize<<<1, 32, 0, stream>>>(acc, out);
}
